// TransformerBlock_7430293422339
// MI455X (gfx1250) — matrix operand order repaired and checked by reading, not compiled, not run on hardware
//
#include <hip/hip_runtime.h>
#include <math.h>

// ---------------- types & helpers -------------------------------------------
typedef __attribute__((ext_vector_type(16))) __bf16 bf16x16;
typedef __attribute__((ext_vector_type(8)))  float  f32x8;

static __device__ inline f32x8 wmma_bf16(bf16x16 a, bf16x16 b, f32x8 c) {
  return __builtin_amdgcn_wmma_f32_16x16x32_bf16(false, a, false, b,
                                                 (short)0, c, false, false);
}

static __device__ inline unsigned short f2bf(float f) {
  unsigned int u = __builtin_bit_cast(unsigned int, f);
  unsigned int r = u + 0x7fffu + ((u >> 16) & 1u);   // RNE
  return (unsigned short)(r >> 16);
}
static __device__ inline __bf16 us2bf(unsigned short s) {
  return __builtin_bit_cast(__bf16, s);
}

// CDNA5 async global->LDS copy, 16B per lane, tracked by ASYNCcnt.
// lds_off = workgroup-relative LDS byte address (low 32 bits of generic ptr).
static __device__ inline void async_copy_b128(unsigned lds_off, const void* g) {
  asm volatile("global_load_async_to_lds_b128 %0, %1, off"
               :: "v"(lds_off), "v"((unsigned long long)(size_t)g)
               : "memory");
}
static __device__ inline void wait_async_all() {
  asm volatile("s_wait_asynccnt 0x0" ::: "memory");
}
static __device__ inline void wait_async_8() {
  asm volatile("s_wait_asynccnt 0x8" ::: "memory");
}

// A-fragment (16x32 bf16, M x K) from a row-major ushort tile (stride ld).
static __device__ inline bf16x16 frag_a(const unsigned short* base, int ld) {
  const int lane = threadIdx.x & 31;
  const int m = lane & 15, hf = lane >> 4;
  bf16x16 a;
#pragma unroll
  for (int r = 0; r < 8; ++r) {
    const int kb = (hf << 4) + (r << 1);
    unsigned int pr = *(const unsigned int*)(base + m * ld + kb);
    a[2 * r]     = us2bf((unsigned short)(pr & 0xffffu));
    a[2 * r + 1] = us2bf((unsigned short)(pr >> 16));
  }
  return a;
}

// B-fragment (32x16 bf16, K x N) from row-major ushort tile (stride ld),
// base already offset to column 0 of the 16-wide N subtile.
static __device__ inline bf16x16 frag_b(const unsigned short* base, int ld) {
  const int lane = threadIdx.x & 31;
  const int n = lane & 15, kh = lane >> 4;
  bf16x16 b;
#pragma unroll
  for (int r = 0; r < 8; ++r) {
    const int k = (kh << 4) + (r << 1);
    b[2 * r]     = us2bf(base[k * ld + n]);
    b[2 * r + 1] = us2bf(base[(k + 1) * ld + n]);
  }
  return b;
}

// ---------------- elementwise kernels ---------------------------------------
__global__ __launch_bounds__(256) void cvt_kernel(const float* __restrict__ in,
                                                  unsigned short* __restrict__ out,
                                                  size_t n) {
  size_t i = (size_t)blockIdx.x * 256 + threadIdx.x;
  if (i < n) out[i] = f2bf(in[i]);
}

__global__ __launch_bounds__(256) void silu_mul_kernel(const float* __restrict__ u,
                                                       const float* __restrict__ g,
                                                       unsigned short* __restrict__ o,
                                                       size_t n) {
  size_t i = (size_t)blockIdx.x * 256 + threadIdx.x;
  if (i < n) {
    float uv = u[i];
    float s = uv / (1.f + __expf(-uv));
    o[i] = f2bf(s * g[i]);
  }
}

__global__ __launch_bounds__(256) void rmsnorm_kernel(const float* __restrict__ x,
                                                      const float* __restrict__ w,
                                                      unsigned short* __restrict__ out) {
  const int row = blockIdx.x;
  const float* xr = x + (size_t)row * 2048;
  float ss = 0.f;
  for (int i = threadIdx.x; i < 2048; i += 256) {
    float v = xr[i];
    ss += v * v;
  }
#pragma unroll
  for (int m = 16; m >= 1; m >>= 1) ss += __shfl_xor(ss, m, 32);
  __shared__ float red[8];
  if ((threadIdx.x & 31) == 0) red[threadIdx.x >> 5] = ss;
  __syncthreads();
  float tot = 0.f;
#pragma unroll
  for (int i = 0; i < 8; ++i) tot += red[i];
  const float sc = rsqrtf(tot * (1.f / 2048.f) + 1e-5f);
  for (int i = threadIdx.x; i < 2048; i += 256)
    out[(size_t)row * 2048 + i] = f2bf(w[i] * xr[i] * sc);
}

__global__ __launch_bounds__(256) void rope_kernel(const float* __restrict__ qf,
                                                   const float* __restrict__ kf,
                                                   const float* __restrict__ cs,
                                                   const float* __restrict__ sn,
                                                   unsigned short* __restrict__ qb,
                                                   unsigned short* __restrict__ kb) {
  const int pos = blockIdx.x;
  for (int i = threadIdx.x; i < 24 * 64; i += 256) {
    const int head = i >> 6, d = i & 63;
    const float c = cs[pos * 64 + d], s = sn[pos * 64 + d];
    if (head < 16) {
      const size_t b = (size_t)pos * 2048 + head * 128 + (d << 1);
      const float xe = qf[b], xo = qf[b + 1];
      qb[b]     = f2bf(xe * c - xo * s);
      qb[b + 1] = f2bf(xe * s + xo * c);
    } else {
      const int kh = head - 16;
      const size_t b = (size_t)pos * 1024 + kh * 128 + (d << 1);
      const float xe = kf[b], xo = kf[b + 1];
      kb[b]     = f2bf(xe * c - xo * s);
      kb[b + 1] = f2bf(xe * s + xo * c);
    }
  }
}

// ---------------- WMMA GEMM: C(f32) = A(bf16) @ B(bf16) [+ residual] ---------
// Block tile 128(M) x 128(N), K step 64, 256 threads = 8 waves (2Mx4N),
// wave tile 64x32 -> 4x2 accumulators, 16 WMMAs per K step.
// Double-buffered LDS filled with global_load_async_to_lds_b128 (ASYNCcnt).
template <bool RES>
__global__ __launch_bounds__(256, 2) void gemm_kernel(
    const unsigned short* __restrict__ A, const unsigned short* __restrict__ B,
    const float* __restrict__ R, float* __restrict__ C, int M, int N, int K) {
  __shared__ alignas(16) unsigned short sA[2][128 * 64];   // 2 x 16 KB
  __shared__ alignas(16) unsigned short sB[2][64 * 128];   // 2 x 16 KB
  const int bm = blockIdx.y << 7;
  const int bn = blockIdx.x << 7;
  const int tid = threadIdx.x;
  const int wid = tid >> 5;
  const int lane = tid & 31;
  const int wm = (wid & 1) << 6;      // 0 / 64
  const int wn = (wid >> 1) << 5;     // 0 / 32 / 64 / 96

  const unsigned aoff[2] = {(unsigned)(size_t)&sA[0][0], (unsigned)(size_t)&sA[1][0]};
  const unsigned boff[2] = {(unsigned)(size_t)&sB[0][0], (unsigned)(size_t)&sB[1][0]};

  f32x8 acc[4][2];
#pragma unroll
  for (int i = 0; i < 4; ++i)
#pragma unroll
    for (int j = 0; j < 2; ++j)
#pragma unroll
      for (int e = 0; e < 8; ++e) acc[i][j][e] = 0.f;

  // stage one K-tile: A 128x64 (4 b128/thread), B 64x128 (4 b128/thread)
  auto stage = [&](int k0, int buf) {
#pragma unroll
    for (int i = 0; i < 4; ++i) {
      const int idx = tid + (i << 8);          // 0..1023
      const int row = idx >> 3, seg = idx & 7; // 8 x 16B segs per 64-col row
      async_copy_b128(aoff[buf] + (unsigned)(row * 128 + seg * 16),
                      A + (size_t)(bm + row) * K + k0 + seg * 8);
    }
#pragma unroll
    for (int i = 0; i < 4; ++i) {
      const int idx = tid + (i << 8);            // 0..1023
      const int row = idx >> 4, seg = idx & 15;  // 16 x 16B segs per 128-col row
      async_copy_b128(boff[buf] + (unsigned)(row * 256 + seg * 16),
                      B + (size_t)(k0 + row) * N + bn + seg * 8);
    }
  };

  const int nsteps = K >> 6;
  stage(0, 0);
  for (int i = 0; i < nsteps; ++i) {
    const int cur = i & 1;
    if (i + 1 < nsteps) {
      stage((i + 1) << 6, cur ^ 1);
      wait_async_8();     // async completes in order: current tile's 8 are done
    } else {
      wait_async_all();
    }
    __syncthreads();

    const unsigned short* cA = sA[cur];
    const unsigned short* cB = sB[cur];
#pragma unroll
    for (int ks = 0; ks < 2; ++ks) {
      bf16x16 af[4];
#pragma unroll
      for (int m = 0; m < 4; ++m)
        af[m] = frag_a(cA + (wm + (m << 4)) * 64 + (ks << 5), 64);
      bf16x16 bfr[2];
#pragma unroll
      for (int nn = 0; nn < 2; ++nn)
        bfr[nn] = frag_b(cB + (ks << 5) * 128 + wn + (nn << 4), 128);
#pragma unroll
      for (int m = 0; m < 4; ++m)
#pragma unroll
        for (int nn = 0; nn < 2; ++nn)
          acc[m][nn] = wmma_bf16(af[m], bfr[nn], acc[m][nn]);
    }
    __syncthreads();   // everyone done reading `cur` before it is re-staged
  }

  const int nl = lane & 15, hf = lane >> 4;
#pragma unroll
  for (int m = 0; m < 4; ++m)
#pragma unroll
    for (int nn = 0; nn < 2; ++nn) {
      const int r0 = bm + wm + (m << 4) + (hf << 3);
      const int c0 = bn + wn + (nn << 4) + nl;
#pragma unroll
      for (int r = 0; r < 8; ++r) {
        const size_t o = (size_t)(r0 + r) * N + c0;
        float v = acc[m][nn][r];
        if (RES) v += R[o];
        C[o] = v;
      }
    }
}

// ---------------- attention helpers -----------------------------------------
static __device__ inline bf16x16 frag_kT(const unsigned short* kb, int key0,
                                         int kvh, int dofs) {
  const int lane = threadIdx.x & 31;
  const int n = lane & 15, kh = lane >> 4;
  const unsigned short* p =
      kb + (size_t)(key0 + n) * 1024 + kvh * 128 + dofs + (kh << 4);
  bf16x16 b;
#pragma unroll
  for (int r = 0; r < 8; ++r) {
    unsigned int pr = *(const unsigned int*)(p + (r << 1));
    b[2 * r]     = us2bf((unsigned short)(pr & 0xffffu));
    b[2 * r + 1] = us2bf((unsigned short)(pr >> 16));
  }
  return b;
}

static __device__ inline bf16x16 frag_v(const unsigned short* vb, int key0,
                                        int kvh, int nt0) {
  const int lane = threadIdx.x & 31;
  const int n = lane & 15, kh = lane >> 4;
  bf16x16 b;
#pragma unroll
  for (int r = 0; r < 8; ++r) {
    const int k = key0 + (kh << 4) + (r << 1);
    b[2 * r]     = us2bf(vb[(size_t)k * 1024 + kvh * 128 + nt0 + n]);
    b[2 * r + 1] = us2bf(vb[(size_t)(k + 1) * 1024 + kvh * 128 + nt0 + n]);
  }
  return b;
}

// Differential flash attention: one wave per (head, 16-query tile).
__global__ __launch_bounds__(32) void attn_kernel(
    const unsigned short* __restrict__ qb, const unsigned short* __restrict__ kb,
    const unsigned short* __restrict__ vb, unsigned short* __restrict__ ob) {
  __shared__ unsigned short sP1[16 * 32];
  __shared__ unsigned short sP2[16 * 32];
  const int h = blockIdx.x;
  const int q0 = blockIdx.y << 4;
  const int kvh = h >> 1;                         // NREP = 2
  const int lane = threadIdx.x & 31;
  const int nl = lane & 15, hf = lane >> 4;

  bf16x16 aq[4];
  {
    const unsigned short* qrow = qb + (size_t)(q0 + nl) * 2048 + h * 128;
#pragma unroll
    for (int c = 0; c < 4; ++c)
#pragma unroll
      for (int r = 0; r < 8; ++r) {
        const int kbi = (c << 5) + (hf << 4) + (r << 1);
        unsigned int pr = *(const unsigned int*)(qrow + kbi);
        aq[c][2 * r]     = us2bf((unsigned short)(pr & 0xffffu));
        aq[c][2 * r + 1] = us2bf((unsigned short)(pr >> 16));
      }
  }

  f32x8 O1[8], O2[8];
  float m1[8], l1[8], m2[8], l2[8];
#pragma unroll
  for (int t = 0; t < 8; ++t) {
#pragma unroll
    for (int e = 0; e < 8; ++e) { O1[t][e] = 0.f; O2[t][e] = 0.f; }
    m1[t] = -1e30f; m2[t] = -1e30f; l1[t] = 0.f; l2[t] = 0.f;
  }

  const int jend = (q0 + 47) >> 5;
  for (int j = 0; j < jend; ++j) {
    const int key0 = j << 5;
    f32x8 S1[2], S2[2];
#pragma unroll
    for (int t = 0; t < 2; ++t) {
      f32x8 z1, z2;
#pragma unroll
      for (int e = 0; e < 8; ++e) { z1[e] = 0.f; z2[e] = 0.f; }
      z1 = wmma_bf16(aq[0], frag_kT(kb, key0 + (t << 4), kvh, 0), z1);
      z1 = wmma_bf16(aq[1], frag_kT(kb, key0 + (t << 4), kvh, 32), z1);
      z2 = wmma_bf16(aq[2], frag_kT(kb, key0 + (t << 4), kvh, 64), z2);
      z2 = wmma_bf16(aq[3], frag_kT(kb, key0 + (t << 4), kvh, 96), z2);
      S1[t] = z1; S2[t] = z2;
    }
    float a1v[8], a2v[8];
#pragma unroll
    for (int r = 0; r < 8; ++r) {
      const int qrow = q0 + (hf << 3) + r;
      float mx1 = -1e30f, mx2 = -1e30f;
#pragma unroll
      for (int t = 0; t < 2; ++t) {
        const int key = key0 + (t << 4) + nl;
        float s1 = S1[t][r] * 0.125f;             // 1/sqrt(64)
        float s2 = S2[t][r] * 0.125f;
        if (key > qrow) { s1 = -1e30f; s2 = -1e30f; }
        S1[t][r] = s1; S2[t][r] = s2;
        mx1 = fmaxf(mx1, s1); mx2 = fmaxf(mx2, s2);
      }
#pragma unroll
      for (int m = 1; m < 16; m <<= 1) {
        mx1 = fmaxf(mx1, __shfl_xor(mx1, m, 32));
        mx2 = fmaxf(mx2, __shfl_xor(mx2, m, 32));
      }
      const float nm1 = fmaxf(m1[r], mx1);
      const float nm2 = fmaxf(m2[r], mx2);
      const float al1 = __expf(m1[r] - nm1);
      const float al2 = __expf(m2[r] - nm2);
      float rs1 = 0.f, rs2 = 0.f;
#pragma unroll
      for (int t = 0; t < 2; ++t) {
        const float p1 = __expf(S1[t][r] - nm1);
        const float p2 = __expf(S2[t][r] - nm2);
        sP1[((hf << 3) + r) * 32 + (t << 4) + nl] = f2bf(p1);
        sP2[((hf << 3) + r) * 32 + (t << 4) + nl] = f2bf(p2);
        rs1 += p1; rs2 += p2;
      }
#pragma unroll
      for (int m = 1; m < 16; m <<= 1) {
        rs1 += __shfl_xor(rs1, m, 32);
        rs2 += __shfl_xor(rs2, m, 32);
      }
      l1[r] = l1[r] * al1 + rs1; m1[r] = nm1; a1v[r] = al1;
      l2[r] = l2[r] * al2 + rs2; m2[r] = nm2; a2v[r] = al2;
    }
#pragma unroll
    for (int nt = 0; nt < 8; ++nt)
#pragma unroll
      for (int r = 0; r < 8; ++r) { O1[nt][r] *= a1v[r]; O2[nt][r] *= a2v[r]; }
    __syncthreads();
    bf16x16 p1f = frag_a(sP1, 32);
    bf16x16 p2f = frag_a(sP2, 32);
#pragma unroll
    for (int nt = 0; nt < 8; ++nt) {
      bf16x16 bv = frag_v(vb, key0, kvh, nt << 4);
      O1[nt] = wmma_bf16(p1f, bv, O1[nt]);
      O2[nt] = wmma_bf16(p2f, bv, O2[nt]);
    }
    __syncthreads();
  }
#pragma unroll
  for (int nt = 0; nt < 8; ++nt)
#pragma unroll
    for (int r = 0; r < 8; ++r) {
      const float o = O1[nt][r] / l1[r] - 0.2f * O2[nt][r] / l2[r];
      const size_t row = q0 + (hf << 3) + r;
      ob[row * 2048 + h * 128 + (nt << 4) + nl] = f2bf(o);
    }
}

// ---------------- host orchestration ----------------------------------------
extern "C" void kernel_launch(void* const* d_in, const int* in_sizes, int n_in,
                              void* d_out, int out_size, void* d_ws, size_t ws_size,
                              hipStream_t stream) {
  const float* x   = (const float*)d_in[0];
  const float* fc  = (const float*)d_in[1];
  const float* fs  = (const float*)d_in[2];
  const float* anw = (const float*)d_in[3];
  const float* wq  = (const float*)d_in[4];
  const float* wk  = (const float*)d_in[5];
  const float* wv  = (const float*)d_in[6];
  const float* wo  = (const float*)d_in[7];
  const float* fnw = (const float*)d_in[8];
  const float* w1  = (const float*)d_in[9];
  const float* w2  = (const float*)d_in[10];
  const float* w3  = (const float*)d_in[11];
  float* outp = (float*)d_out;

  const int L = 2048, D = 2048, KV = 1024, HID = 5632;

  char* base = (char*)d_ws;
  size_t off = 0;
  auto alloc = [&](size_t bytes) -> void* {
    void* p = base + off;
    off = (off + bytes + 255) & ~(size_t)255;
    return p;
  };
  unsigned short* wq_b = (unsigned short*)alloc((size_t)D * D * 2);
  unsigned short* wk_b = (unsigned short*)alloc((size_t)D * KV * 2);
  unsigned short* wv_b = (unsigned short*)alloc((size_t)D * KV * 2);
  unsigned short* wo_b = (unsigned short*)alloc((size_t)D * D * 2);
  unsigned short* w1_b = (unsigned short*)alloc((size_t)D * HID * 2);
  unsigned short* w2_b = (unsigned short*)alloc((size_t)HID * D * 2);
  unsigned short* w3_b = (unsigned short*)alloc((size_t)D * HID * 2);
  unsigned short* h_b  = (unsigned short*)alloc((size_t)L * D * 2);
  float* qf = (float*)alloc((size_t)L * D * 4);
  float* kf = (float*)alloc((size_t)L * KV * 4);
  float* vf = (float*)alloc((size_t)L * KV * 4);
  unsigned short* q_b  = (unsigned short*)alloc((size_t)L * D * 2);
  unsigned short* k_b  = (unsigned short*)alloc((size_t)L * KV * 2);
  unsigned short* v_b  = (unsigned short*)alloc((size_t)L * KV * 2);
  unsigned short* at_b = (unsigned short*)alloc((size_t)L * D * 2);
  float* h2 = (float*)alloc((size_t)L * D * 4);
  unsigned short* f_b  = (unsigned short*)alloc((size_t)L * D * 2);
  float* uf = (float*)alloc((size_t)L * HID * 4);
  float* gf = (float*)alloc((size_t)L * HID * 4);
  unsigned short* act_b = (unsigned short*)alloc((size_t)L * HID * 2);

  auto cvt = [&](const float* s, unsigned short* d, size_t n) {
    cvt_kernel<<<dim3((unsigned)((n + 255) / 256)), 256, 0, stream>>>(s, d, n);
  };
  cvt(wq, wq_b, (size_t)D * D);
  cvt(wk, wk_b, (size_t)D * KV);
  cvt(wv, wv_b, (size_t)D * KV);
  cvt(wo, wo_b, (size_t)D * D);
  cvt(w1, w1_b, (size_t)D * HID);
  cvt(w2, w2_b, (size_t)HID * D);
  cvt(w3, w3_b, (size_t)D * HID);

  rmsnorm_kernel<<<L, 256, 0, stream>>>(x, anw, h_b);

  gemm_kernel<false><<<dim3(D / 128, L / 128), 256, 0, stream>>>(h_b, wq_b, nullptr, qf, L, D, D);
  gemm_kernel<false><<<dim3(KV / 128, L / 128), 256, 0, stream>>>(h_b, wk_b, nullptr, kf, L, KV, D);
  gemm_kernel<false><<<dim3(KV / 128, L / 128), 256, 0, stream>>>(h_b, wv_b, nullptr, vf, L, KV, D);

  rope_kernel<<<L, 256, 0, stream>>>(qf, kf, fc, fs, q_b, k_b);
  cvt(vf, v_b, (size_t)L * KV);

  attn_kernel<<<dim3(16, L / 16), 32, 0, stream>>>(q_b, k_b, v_b, at_b);

  gemm_kernel<true><<<dim3(D / 128, L / 128), 256, 0, stream>>>(at_b, wo_b, x, h2, L, D, D);

  rmsnorm_kernel<<<L, 256, 0, stream>>>(h2, fnw, f_b);

  gemm_kernel<false><<<dim3(HID / 128, L / 128), 256, 0, stream>>>(f_b, w1_b, nullptr, uf, L, HID, D);
  gemm_kernel<false><<<dim3(HID / 128, L / 128), 256, 0, stream>>>(f_b, w3_b, nullptr, gf, L, HID, D);
  silu_mul_kernel<<<dim3((unsigned)(((size_t)L * HID + 255) / 256)), 256, 0, stream>>>(
      uf, gf, act_b, (size_t)L * HID);

  gemm_kernel<true><<<dim3(D / 128, L / 128), 256, 0, stream>>>(act_b, w2_b, h2, outp, L, D, HID);
}
